// LearnablePixelwiseAnisoJBU_NoParent_39127152066849
// MI455X (gfx1250) — matrix operand order repaired and checked by reading, not compiled, not run on hardware
//
#include <hip/hip_runtime.h>
#include <hip/hip_bf16.h>
#include <math.h>

// ---------------------------------------------------------------------------
// LearnablePixelwiseAnisoJBU on MI455X (gfx1250, wave32, WMMA)
//
// Structure: SCALE=16 => each 16x16 HR tile shares one LR center cell
// (uc=ty, vc=tx), so its 169-cell footprint is tile-uniform and the whole
// tile is an exact dense GEMM:  out_tile[256 pix x 256 ch] =
//     W[256 x 192(=169 padded)] @ F[192 x 256],  then scale rows by 1/den.
// GEMM runs on v_wmma_f32_16x16x32_f16 (f16 in, f32 accumulate).
// A-fragments are hoisted out of the N-tile loop (each wave owns 2 M-tiles),
// roughly halving LDS fragment traffic in the GEMM phase.
// ---------------------------------------------------------------------------

typedef _Float16 v16h __attribute__((ext_vector_type(16)));
typedef _Float16 v8h  __attribute__((ext_vector_type(8)));
typedef float    v8f  __attribute__((ext_vector_type(8)));

#define HL 20
#define WL 20
#define NC 256
#define HH 320
#define WH 320
#define PIX (HH*WH)      // 102400
#define NCELL (HL*WL)    // 400
#define KNB 169          // 13*13 neighbor taps
#define KPAD 192         // K padded to multiple of 32 for WMMA
#define WSTRIDE 200      // halves per W row (192 + 8 pad -> conflict-free frags)
#define BSTRIDE 264      // halves per B row (256 + 8 pad)

struct CellParam {  // 48 B
    float cs, sn, i2sx2, i2sy2, i2sr2, g0, g1, g2, se, pad0, pad1, pad2;
};

// workspace layout
#define WS_CELLS_OFF 0
#define WS_FEATT_OFF 19456   // 400*48=19200, rounded up to 1024

// dynamic LDS layout (bytes)
#define SM_W    0
#define SM_B    (NC*WSTRIDE*2)              // 102400
#define SM_P0   (SM_B + KPAD*BSTRIDE*2)     // 203776
#define SM_P1   (SM_P0 + 176*16)            // 206592
#define SM_P2   (SM_P1 + 176*16)            // 209408
#define SM_CIDX (SM_P2 + 176*16)            // 212224
#define SM_IDEN (SM_CIDX + 176*4)           // 212928
#define SM_STG  (SM_IDEN + 256*4)           // 213952
#define SM_TOTAL (SM_STG + 8*16*17*4)       // 222656  (< 320 KB per WGP)

// ---------------------------------------------------------------------------
// Prep 1: per-LR-cell parameters (exp/tanh/trig, inverse variances,
// bilinearly downsampled guide, sigma_eff for the R_map bilerp).
// ---------------------------------------------------------------------------
__global__ void prep_cells(const float* __restrict__ guide,
                           const float* __restrict__ sx_raw,
                           const float* __restrict__ sy_raw,
                           const float* __restrict__ th_raw,
                           const float* __restrict__ sr_raw,
                           CellParam* __restrict__ cells) {
    int cell = blockIdx.x * blockDim.x + threadIdx.x;
    if (cell >= NCELL) return;
    int i = cell / WL, j = cell % WL;

    float sx = __expf(sx_raw[cell]);
    float sy = __expf(sy_raw[cell]);
    float th = 3.14159265358979323846f * tanhf(th_raw[cell]);
    float sr = __expf(sr_raw[cell]);

    CellParam cp;
    cp.se = fmaxf(sx, sy);
    float sxc = fmaxf(sx, 1e-6f), syc = fmaxf(sy, 1e-6f), src = fmaxf(sr, 1e-6f);
    cp.cs = __cosf(th);
    cp.sn = __sinf(th);
    cp.i2sx2 = 1.0f / (2.0f*sxc*sxc + 1e-8f);
    cp.i2sy2 = 1.0f / (2.0f*syc*syc + 1e-8f);
    cp.i2sr2 = 1.0f / (2.0f*src*src + 1e-8f);

    // guide_lr = bilerp(guide_hr, 20, 20), align_corners=False convention
    float syy = fminf(fmaxf(((float)i + 0.5f)*16.0f - 0.5f, 0.0f), (float)(HH-1));
    float sxx = fminf(fmaxf(((float)j + 0.5f)*16.0f - 0.5f, 0.0f), (float)(WH-1));
    int y0 = (int)floorf(syy); int y1 = min(y0 + 1, HH-1); float ly = syy - (float)y0;
    int x0 = (int)floorf(sxx); int x1 = min(x0 + 1, WH-1); float lx = sxx - (float)x0;
    float g[3];
#pragma unroll
    for (int c = 0; c < 3; ++c) {
        const float* b = guide + (size_t)c * PIX;
        float v00 = b[y0*WH + x0], v01 = b[y0*WH + x1];
        float v10 = b[y1*WH + x0], v11 = b[y1*WH + x1];
        g[c] = (1.0f-ly)*((1.0f-lx)*v00 + lx*v01) + ly*((1.0f-lx)*v10 + lx*v11);
    }
    cp.g0 = g[0]; cp.g1 = g[1]; cp.g2 = g[2];
    cp.pad0 = cp.pad1 = cp.pad2 = 0.0f;
    cells[cell] = cp;
}

// ---------------------------------------------------------------------------
// Prep 2: feat [C=256][400] f32  ->  featT [cell=400][256] f16
// (coalesced per-footprint-row gathers in the main kernel, half the bytes)
// ---------------------------------------------------------------------------
__global__ void prep_featT(const float* __restrict__ feat,
                           _Float16* __restrict__ featT) {
    int t = blockIdx.x * blockDim.x + threadIdx.x;
    if (t >= NCELL * NC) return;
    int c = t / NCELL, cell = t % NCELL;          // feat read is contiguous
    featT[(size_t)cell * NC + c] = (_Float16)feat[t];
}

// ---------------------------------------------------------------------------
// Main kernel: one 16x16 HR tile per workgroup (grid 20x20, 256 threads).
// ---------------------------------------------------------------------------
__global__ void __launch_bounds__(256)
jbu_main(const CellParam* __restrict__ cells,
         const _Float16* __restrict__ featT,
         const float* __restrict__ guide,
         float* __restrict__ out) {
    extern __shared__ char smem[];
    _Float16* Wlds  = (_Float16*)(smem + SM_W);
    _Float16* Blds  = (_Float16*)(smem + SM_B);
    float4*   P0    = (float4*)(smem + SM_P0);   // {cy, cx, cos, sin}
    float4*   P1    = (float4*)(smem + SM_P1);   // {i2sx2, i2sy2, i2sr2, rad2}
    float4*   P2    = (float4*)(smem + SM_P2);   // {g0, g1, g2, 0}
    int*      cidx  = (int*)(smem + SM_CIDX);
    float*    invden= (float*)(smem + SM_IDEN);
    float*    stg   = (float*)(smem + SM_STG);

    const int tid = threadIdx.x;
    const int tx = blockIdx.x, ty = blockIdx.y;

    // ---- Phase A: footprint cell table (tile-uniform) --------------------
    if (tid < KNB) {
        int dyy = tid / 13 - 6, dxx = tid % 13 - 6;
        int Ui = min(max(ty + dyy, 0), HL-1);
        int Vi = min(max(tx + dxx, 0), WL-1);
        int cell = Ui * WL + Vi;
        cidx[tid] = cell;
        CellParam cp = cells[cell];
        float cy = ((float)Ui + 0.5f)*16.0f - 0.5f;
        float cx = ((float)Vi + 0.5f)*16.0f - 0.5f;
        P0[tid] = make_float4(cy, cx, cp.cs, cp.sn);
        P1[tid] = make_float4(cp.i2sx2, cp.i2sy2, cp.i2sr2,
                              (float)(dyy*dyy + dxx*dxx));
        P2[tid] = make_float4(cp.g0, cp.g1, cp.g2, 0.0f);
    }
    __syncthreads();

    // ---- Phase B: stage F matrix [KPAD x 256] f16 into LDS ---------------
    {
        int rsub = tid >> 6;            // 4 rows per iteration
        int c4   = (tid & 63) * 4;      // 4 halves (8 B) per thread
#pragma unroll 4
        for (int kq = 0; kq < 48; ++kq) {
            int k2 = kq * 4 + rsub;
            uint2 v = make_uint2(0u, 0u);          // zero-fill K pad rows
            if (k2 < KNB)
                v = *(const uint2*)(featT + (size_t)cidx[k2] * NC + c4);
            *(uint2*)(Blds + k2 * BSTRIDE + c4) = v;
        }
    }

    // ---- Weights: thread == pixel; two passes (max, then exp) ------------
    const int py = tid >> 4, px = tid & 15;
    const int Y = ty*16 + py, X = tx*16 + px;
    const float fY = (float)Y, fX = (float)X;
    const float gp0 = guide[0*PIX + Y*WH + X];
    const float gp1 = guide[1*PIX + Y*WH + X];
    const float gp2 = guide[2*PIX + Y*WH + X];

    // R_map = clip(ceil(2 * bilerp(sigma_eff)), 1, 6)
    float R2;
    {
        float syy = fminf(fmaxf((fY + 0.5f)/16.0f - 0.5f, 0.0f), (float)(HL-1));
        float sxx = fminf(fmaxf((fX + 0.5f)/16.0f - 0.5f, 0.0f), (float)(WL-1));
        int y0 = (int)floorf(syy); int y1 = min(y0 + 1, HL-1); float ly = syy - (float)y0;
        int x0 = (int)floorf(sxx); int x1 = min(x0 + 1, WL-1); float lx = sxx - (float)x0;
        float v00 = cells[y0*WL + x0].se, v01 = cells[y0*WL + x1].se;
        float v10 = cells[y1*WL + x0].se, v11 = cells[y1*WL + x1].se;
        float se = (1.0f-ly)*((1.0f-lx)*v00 + lx*v01) + ly*((1.0f-lx)*v10 + lx*v11);
        float Rm = fminf(fmaxf(ceilf(2.0f*se), 1.0f), 6.0f);
        R2 = Rm * Rm;
    }

    _Float16* wrow = Wlds + tid * WSTRIDE;
    float mmax = -INFINITY;
    for (int k = 0; k < KNB; ++k) {
        float4 p0 = P0[k], p1 = P1[k], p2 = P2[k];
        float dyp = fY - p0.x, dxp = fX - p0.y;
        float xp = dxp*p0.z + dyp*p0.w;
        float yp = dyp*p0.z - dxp*p0.w;
        float lw = -(xp*xp*p1.x + yp*yp*p1.y);
        float d0 = gp0 - p2.x, d1 = gp1 - p2.y, d2 = gp2 - p2.z;
        lw -= (d0*d0 + d1*d1 + d2*d2) * p1.z;
        lw = (p1.w <= R2) ? lw : -INFINITY;      // radius mask
        mmax = fmaxf(mmax, lw);
        wrow[k] = (_Float16)lw;
    }
    float den = 0.0f;
    for (int k = 0; k < KNB; ++k) {
        float e = __expf((float)wrow[k] - mmax);
        den += e;
        wrow[k] = (_Float16)e;
    }
    for (int k = KNB; k < KPAD; ++k) wrow[k] = (_Float16)0.0f;
    // Center tap (rad2=0 <= R^2>=1) is always unmasked => den >= 1, so the
    // reference's den<1e-6 bilinear fallback and max(den,1e-8) are dead code.
    invden[tid] = 1.0f / den;
    __syncthreads();

    // ---- GEMM: 16x16 M-tiles x 16 N-tiles, 8 waves x (2 M x 16 N) --------
    const int wave = tid >> 5, lane = tid & 31;
    const int laneHi = lane >> 4;       // 0: K lo-half / M rows 0-7
    const int lane15 = lane & 15;
    float* wstg = stg + wave * (16*17);

#pragma unroll 1
    for (int mh = 0; mh < 2; ++mh) {
        const int mt = wave*2 + mh;         // pixel-row tile == HR row py=mt

        // Hoist all 6 A fragments for this M-tile (48 VGPRs, reused 16x).
        // A frag: lane(0-15)=row, K {base..+7, base+16..+23}, base=8*laneHi
        union { v16h v; v8h h[2]; } afr[6];
        {
            const _Float16* arow = Wlds + (mt*16 + lane15)*WSTRIDE + laneHi*8;
#pragma unroll
            for (int kk = 0; kk < 6; ++kk) {
                afr[kk].h[0] = *(const v8h*)(arow + kk*32);
                afr[kk].h[1] = *(const v8h*)(arow + kk*32 + 16);
            }
        }
        const float invd = invden[mt*16 + lane15];
        float* obase = out + (size_t)(ty*16 + mt)*WH + tx*16 + lane15;

#pragma unroll 1
        for (int nt = 0; nt < 16; ++nt) {
            v8f acc = {0.f,0.f,0.f,0.f,0.f,0.f,0.f,0.f};
            // B frag: lane(0-15)=col N, K {base..+7, base+16..+23}, base=8*laneHi
            const _Float16* bcol = Blds + laneHi*8*BSTRIDE + nt*16 + lane15;
#pragma unroll
            for (int kk = 0; kk < 6; ++kk) {
                union { v16h v; v8h h[2]; } b;
#pragma unroll
                for (int s = 0; s < 16; ++s)
                    b.v[s] = bcol[(kk*32 + (s < 8 ? s : s + 8))*BSTRIDE];
                acc = __builtin_amdgcn_wmma_f32_16x16x32_f16(
                          false, afr[kk].v, false, b.v, (short)0, acc,
                          false, false);
            }

            // stage D (lane: M = r + 8*laneHi, N = lane15) via LDS transpose
#pragma unroll
            for (int r = 0; r < 8; ++r)
                wstg[(r + laneHi*8)*17 + lane15] = acc[r];
            asm volatile("s_wait_dscnt 0x0" ::: "memory");

            // write: M == px within the tile -> 64B-contiguous per half-wave
#pragma unroll
            for (int j = 0; j < 8; ++j) {
                int cl = j*2 + laneHi;
                obase[(size_t)(nt*16 + cl) * PIX] = wstg[lane15*17 + cl] * invd;
            }
            asm volatile("s_wait_dscnt 0x0" ::: "memory");  // WAR on wstg
        }
    }
}

// ---------------------------------------------------------------------------
extern "C" void kernel_launch(void* const* d_in, const int* in_sizes, int n_in,
                              void* d_out, int out_size, void* d_ws, size_t ws_size,
                              hipStream_t stream) {
    (void)in_sizes; (void)n_in; (void)out_size; (void)ws_size;
    const float* feat  = (const float*)d_in[0];   // [1,256,20,20]
    const float* guide = (const float*)d_in[1];   // [1,3,320,320]
    const float* sxr   = (const float*)d_in[2];
    const float* syr   = (const float*)d_in[3];
    const float* thr   = (const float*)d_in[4];
    const float* srr   = (const float*)d_in[5];
    float* out = (float*)d_out;

    CellParam* cells = (CellParam*)((char*)d_ws + WS_CELLS_OFF);
    _Float16*  featT = (_Float16*)((char*)d_ws + WS_FEATT_OFF);

    prep_cells<<<2, 256, 0, stream>>>(guide, sxr, syr, thr, srr, cells);
    prep_featT<<<(NCELL*NC + 255)/256, 256, 0, stream>>>(feat, featT);

    hipFuncSetAttribute((const void*)jbu_main,
                        hipFuncAttributeMaxDynamicSharedMemorySize, SM_TOTAL);
    jbu_main<<<dim3(WL, HL), 256, SM_TOTAL, stream>>>(cells, featT, guide, out);
}
